// RetentionMixerModule_76295799046722
// MI455X (gfx1250) — matrix operand order repaired and checked by reading, not compiled, not run on hardware
//
#include <hip/hip_runtime.h>

typedef __bf16 bf16_t;
typedef __attribute__((ext_vector_type(16))) __bf16 v16bf;
typedef __attribute__((ext_vector_type(8)))  float   v8f;
typedef __attribute__((ext_vector_type(4)))  int     v4i;

#define BATCH 4
#define SEQ   4096
#define DIM   2048

#define GEMM_M (BATCH * SEQ)   // 16384
#define GEMM_N DIM             // 2048
#define GEMM_K DIM             // 2048

#define BM 128
#define BN 128
#define BK 32
#define LDT 40                 // padded LDS row stride (elements): 80B -> conflict-free
#define NKT (GEMM_K / BK)      // 64

#if __has_builtin(__builtin_amdgcn_global_load_async_to_lds_b128)
#define USE_ASYNC_LDS 1
#else
#define USE_ASYNC_LDS 0
#endif

typedef __attribute__((address_space(1))) v4i* gv4i_p;   // global int4*
typedef __attribute__((address_space(3))) v4i* lv4i_p;   // LDS int4*
typedef __attribute__((address_space(3))) bf16_t lds_bf16;

__device__ __forceinline__ void wait_async0() {
#if __has_builtin(__builtin_amdgcn_s_wait_asynccnt)
  __builtin_amdgcn_s_wait_asynccnt(0);
#else
  asm volatile("s_wait_asynccnt 0x0" ::: "memory");
#endif
}

union FragU { v16bf v; uint4 q[2]; };

// A fragment (16x32 bf16): lanes 0-15 hold K{0..7,16..23}, lanes 16-31 K{8..15,24..31}
__device__ __forceinline__ v16bf load_frag_a(const bf16_t* base) {
  FragU u;
  u.q[0] = *(const uint4*)(base);        // 8 bf16
  u.q[1] = *(const uint4*)(base + 16);   // 8 bf16, +16 elements
  return u.v;
}
// B fragment (32x16 bf16): lanes 0-15 hold K{0..7,16..23}, lanes 16-31 K{8..15,24..31}
__device__ __forceinline__ v16bf load_frag_b(const bf16_t* base) {
  FragU u;
  u.q[0] = *(const uint4*)(base);
  u.q[1] = *(const uint4*)(base + 16);
  return u.v;
}

// C[M,N] = A[M,K] @ B[N,K]^T   (B row-major [N,K] == torch Linear weight)
template <bool STORE_BF16>
__global__ __launch_bounds__(256, 2)
void wmma_gemm_kernel(const bf16_t* __restrict__ A,
                      const bf16_t* __restrict__ Bw,
                      bf16_t* __restrict__ outB,
                      float*  __restrict__ outF,
                      const float* __restrict__ bias)
{
  __shared__ bf16_t As[2][BM * LDT];
  __shared__ bf16_t Bs[2][BM * LDT];

  const int tid   = threadIdx.x;
  const int lane  = tid & 31;
  const int wid   = tid >> 5;      // 0..7
  const int waveM = wid >> 2;      // 0..1 -> 64 rows each
  const int waveN = wid & 3;       // 0..3 -> 32 cols each
  const int lhalf = lane >> 4;     // 0 or 1
  const int l16   = lane & 15;

  const int mBlock = blockIdx.x * BM;
  const int nBlock = blockIdx.y * BN;

  // global->LDS staging: each thread moves 32B of A and 32B of B per stage
  const int ldRow = tid >> 1;          // 0..127
  const int ldCol = (tid & 1) << 4;    // 0 or 16

  const bf16_t* aG = A  + (size_t)(mBlock + ldRow) * GEMM_K + ldCol;
  const bf16_t* bG = Bw + (size_t)(nBlock + ldRow) * GEMM_K + ldCol;
  bf16_t* aS = &As[0][0] + ldRow * LDT + ldCol;
  bf16_t* bS = &Bs[0][0] + ldRow * LDT + ldCol;

  const bf16_t* aF = &As[0][0] + (waveM * 64 + l16) * LDT + (lhalf << 3); // +0 / +8
  const bf16_t* bF = &Bs[0][0] + (waveN * 32 + l16) * LDT + (lhalf << 3); // +0 / +8

  v8f acc[4][2] = {};

#if USE_ASYNC_LDS
  lds_bf16* aSl = (lds_bf16*)aS;
  lds_bf16* bSl = (lds_bf16*)bS;
  // Direct global -> LDS DMA, tracked with ASYNCcnt (no VGPR staging).
  #define STAGE_TILE(bufv, ktv)                                                          \
    do {                                                                                 \
      const bf16_t* ag_ = aG + (size_t)(ktv) * BK;                                       \
      const bf16_t* bg_ = bG + (size_t)(ktv) * BK;                                       \
      lds_bf16* as_ = aSl + (bufv) * (BM * LDT);                                         \
      lds_bf16* bs_ = bSl + (bufv) * (BM * LDT);                                         \
      __builtin_amdgcn_global_load_async_to_lds_b128((gv4i_p)ag_,       (lv4i_p)as_,       0, 0); \
      __builtin_amdgcn_global_load_async_to_lds_b128((gv4i_p)(ag_ + 8), (lv4i_p)(as_ + 8), 0, 0); \
      __builtin_amdgcn_global_load_async_to_lds_b128((gv4i_p)bg_,       (lv4i_p)bs_,       0, 0); \
      __builtin_amdgcn_global_load_async_to_lds_b128((gv4i_p)(bg_ + 8), (lv4i_p)(bs_ + 8), 0, 0); \
    } while (0)

  STAGE_TILE(0, 0);
  wait_async0();
  __syncthreads();
#else
  // Fallback: register staging
  uint4 ra0 = *(const uint4*)(aG);
  uint4 ra1 = *(const uint4*)(aG + 8);
  uint4 rb0 = *(const uint4*)(bG);
  uint4 rb1 = *(const uint4*)(bG + 8);
  *(uint4*)(aS)     = ra0;
  *(uint4*)(aS + 8) = ra1;
  *(uint4*)(bS)     = rb0;
  *(uint4*)(bS + 8) = rb1;
  __syncthreads();
#endif

  for (int kt = 0; kt < NKT; ++kt) {
    const int cur = kt & 1;

#if USE_ASYNC_LDS
    if (kt + 1 < NKT) STAGE_TILE((kt + 1) & 1, kt + 1);
#else
    if (kt + 1 < NKT) {  // prefetch next K-slab into registers
      const bf16_t* ag = aG + (size_t)(kt + 1) * BK;
      const bf16_t* bg = bG + (size_t)(kt + 1) * BK;
      ra0 = *(const uint4*)(ag);
      ra1 = *(const uint4*)(ag + 8);
      rb0 = *(const uint4*)(bg);
      rb1 = *(const uint4*)(bg + 8);
    }
#endif

    const bf16_t* aFc = aF + cur * (BM * LDT);
    const bf16_t* bFc = bF + cur * (BM * LDT);

    v16bf af[4];
    #pragma unroll
    for (int i = 0; i < 4; ++i) af[i] = load_frag_a(aFc + i * 16 * LDT);
    v16bf bfr[2];
    #pragma unroll
    for (int j = 0; j < 2; ++j) bfr[j] = load_frag_b(bFc + j * 16 * LDT);

    #pragma unroll
    for (int i = 0; i < 4; ++i)
      #pragma unroll
      for (int j = 0; j < 2; ++j)
        acc[i][j] = __builtin_amdgcn_wmma_f32_16x16x32_bf16(
            false, af[i], false, bfr[j], (short)0, acc[i][j], false, false);

    if (kt + 1 < NKT) {
#if USE_ASYNC_LDS
      wait_async0();
      __syncthreads();
#else
      const int nxt = (kt + 1) & 1;
      bf16_t* as = aS + nxt * (BM * LDT);
      bf16_t* bs = bS + nxt * (BM * LDT);
      *(uint4*)(as)     = ra0;
      *(uint4*)(as + 8) = ra1;
      *(uint4*)(bs)     = rb0;
      *(uint4*)(bs + 8) = rb1;
      __syncthreads();
#endif
    }
  }

  // D layout: lane gives N=l16, VGPR v gives M = v + 8*lhalf
  #pragma unroll
  for (int i = 0; i < 4; ++i) {
    const int rowBase = mBlock + waveM * 64 + i * 16 + (lhalf << 3);
    #pragma unroll
    for (int j = 0; j < 2; ++j) {
      const int col = nBlock + waveN * 32 + j * 16 + l16;
      #pragma unroll
      for (int v = 0; v < 8; ++v) {
        const size_t off = (size_t)(rowBase + v) * GEMM_N + col;
        if constexpr (STORE_BF16) {
          outB[off] = (bf16_t)acc[i][j][v];
        } else {
          outF[off] = acc[i][j][v] + bias[col];
        }
      }
    }
  }
}

__global__ __launch_bounds__(256)
void cvt_f32_bf16_kernel(const float* __restrict__ in, bf16_t* __restrict__ out, int n4) {
  int i = blockIdx.x * blockDim.x + threadIdx.x;
  if (i >= n4) return;
  float4 f = ((const float4*)in)[i];
  union { bf16_t h[4]; uint2 u; } p;
  p.h[0] = (bf16_t)f.x;
  p.h[1] = (bf16_t)f.y;
  p.h[2] = (bf16_t)f.z;
  p.h[3] = (bf16_t)f.w;
  ((uint2*)out)[i] = p.u;
}

// In-place linear recurrence over time:  st = d*st + (1-d)*p, one thread per (b, channel)
__global__ __launch_bounds__(256)
void recurrence_kernel(bf16_t* __restrict__ P, const float* __restrict__ decay) {
  const int idx = blockIdx.x * blockDim.x + threadIdx.x;   // 0 .. BATCH*DIM-1
  const int b = idx >> 11;            // / DIM
  const int c = idx & (DIM - 1);
  const float dg = 1.0f / (1.0f + __expf(-decay[c]));
  const float om = 1.0f - dg;
  bf16_t* p = P + (size_t)b * SEQ * DIM + c;
  float st = 0.0f;
  #pragma unroll 4
  for (int s = 0; s < SEQ; ++s) {
    st = dg * st + om * (float)p[(size_t)s * DIM];
    p[(size_t)s * DIM] = (bf16_t)st;
  }
}

extern "C" void kernel_launch(void* const* d_in, const int* in_sizes, int n_in,
                              void* d_out, int out_size, void* d_ws, size_t ws_size,
                              hipStream_t stream) {
  (void)in_sizes; (void)n_in; (void)out_size; (void)ws_size;

  const float* x     = (const float*)d_in[0];
  const float* W_in  = (const float*)d_in[1];
  const float* W_out = (const float*)d_in[2];
  const float* decay = (const float*)d_in[3];
  const float* bias  = (const float*)d_in[4];

  // workspace layout (bf16): x | W_in | W_out | P (reused in-place by recurrence)
  bf16_t* xb  = (bf16_t*)d_ws;
  bf16_t* wib = xb  + (size_t)GEMM_M * GEMM_K;
  bf16_t* wob = wib + (size_t)GEMM_N * GEMM_K;
  bf16_t* pb  = wob + (size_t)GEMM_N * GEMM_K;

  const int nX4 = (GEMM_M * GEMM_K) / 4;   // 8388608
  const int nW4 = (GEMM_N * GEMM_K) / 4;   // 1048576
  cvt_f32_bf16_kernel<<<nX4 / 256, 256, 0, stream>>>(x,     xb,  nX4);
  cvt_f32_bf16_kernel<<<nW4 / 256, 256, 0, stream>>>(W_in,  wib, nW4);
  cvt_f32_bf16_kernel<<<nW4 / 256, 256, 0, stream>>>(W_out, wob, nW4);

  dim3 grid(GEMM_M / BM, GEMM_N / BN);   // 128 x 16
  wmma_gemm_kernel<true><<<grid, 256, 0, stream>>>(xb, wib, pb, nullptr, nullptr);

  recurrence_kernel<<<(BATCH * DIM) / 256, 256, 0, stream>>>(pb, decay);

  wmma_gemm_kernel<false><<<grid, 256, 0, stream>>>(pb, wob, nullptr, (float*)d_out, bias);
}
